// SlidingWindowDocumentRQVAE_17171279249692
// MI455X (gfx1250) — matrix operand order repaired and checked by reading, not compiled, not run on hardware
//
#include <hip/hip_runtime.h>
#include <hip/hip_fp16.h>

typedef __attribute__((ext_vector_type(16))) _Float16 v16h;
typedef __attribute__((ext_vector_type(8)))  _Float16 v8h;
typedef __attribute__((ext_vector_type(4)))  _Float16 v4h;
typedef __attribute__((ext_vector_type(8)))  float    v8f;
typedef __attribute__((ext_vector_type(4)))  float    v4f;
typedef __attribute__((ext_vector_type(4)))  unsigned int u32x4;
typedef __attribute__((ext_vector_type(8)))  int      i32x8;
typedef __attribute__((ext_vector_type(4)))  int      i32x4;

#define B_   256
#define W_   41
#define D_   512
#define K_   4096
#define L_   8
#define N_   (B_ * W_)        // 10496 rows
#define NQ_  ((long)N_ * D_)  // 5373952 elements of quantized_out
#define ROWTILES (N_ / 16)    // 656 (exact)
#define KTILES   (K_ / 16)    // 256
#define COMMIT 0.25f

// ---------------------------------------------------------------------------
// prep: codebooks f32 -> f16 (for WMMA B operand) + ||e||^2 per code (f32).
// One wave (32 lanes) per codebook row; 8 waves / block; grid = L*K/8.
// ---------------------------------------------------------------------------
__global__ void prep_codebook(const float* __restrict__ cb,
                              _Float16* __restrict__ cbh,
                              float* __restrict__ enorm) {
  const int wave = threadIdx.x >> 5;
  const int lane = threadIdx.x & 31;
  const long row = (long)blockIdx.x * 8 + wave;     // 0 .. L*K-1
  const float* src = cb + row * D_;
  _Float16*    dst = cbh + row * D_;
  float ss = 0.0f;
#pragma unroll
  for (int j = 0; j < 4; ++j) {
    const int c = j * 128 + lane * 4;               // coalesced float4
    v4f v = *(const v4f*)(src + c);
    ss += v.x * v.x + v.y * v.y + v.z * v.z + v.w * v.w;
    v4h h = { (_Float16)v.x, (_Float16)v.y, (_Float16)v.z, (_Float16)v.w };
    *(v4h*)(dst + c) = h;
  }
#pragma unroll
  for (int m = 1; m < 32; m <<= 1) ss += __shfl_xor(ss, m, 32);
  if (lane == 0) enorm[row] = ss;
}

// ---------------------------------------------------------------------------
// init: residual <- x (f32 + f16 staging), zero d_out (incl. loss + indices),
// zero the ssq accumulator.
// ---------------------------------------------------------------------------
__global__ void init_kernel(const float* __restrict__ x,
                            float* __restrict__ resid,
                            _Float16* __restrict__ residh,
                            float* __restrict__ outp, long out_elems,
                            float* __restrict__ ssq) {
  const long i = (long)blockIdx.x * blockDim.x + threadIdx.x;
  if (i < NQ_) {
    float v = x[i];
    resid[i]  = v;
    residh[i] = (_Float16)v;
  }
  if (i < out_elems) outp[i] = 0.0f;
  if (i == 0) *ssq = 0.0f;
}

// ---------------------------------------------------------------------------
// One RVQ layer: per block = one 16-row tile.
//   Phase 1: TDM (tensor_load_to_lds) stages the 16x512 f16 residual tile
//            into LDS with the padded 520-half row stride (fallback: vector
//            loads if the builtin is unavailable).
//   Phase 2: 8 waves sweep 256 code-tiles; 16 WMMAs per tile split across two
//            independent accumulator chains; running argmin of
//            ||e||^2 - 2*x.e per row; wave butterfly + LDS block reduce.
//   Phase 3: gather winning f32 codes, update residual (f32+f16), accumulate
//            quantized_out, record indices, accumulate sum((r-q)^2).
// ---------------------------------------------------------------------------
__global__ __launch_bounds__(256)
void rvq_layer(float* __restrict__ resid,
               _Float16* __restrict__ residh,
               const _Float16* __restrict__ cbh,   // this layer, [K][D] f16
               const float* __restrict__ cbf,      // this layer, [K][D] f32
               const float* __restrict__ enorm,    // this layer, [K]
               float* __restrict__ outq,           // [N*D] accumulator (d_out)
               float* __restrict__ idxout,         // [N*L] (as float, d_out tail)
               float* __restrict__ ssq,
               int layer) {
  __shared__ _Float16 lA[16][520];   // 16 rows x 512 f16, padded stride (+4 dw)
  __shared__ float    wredS[8][16];
  __shared__ int      wredI[8][16];
  __shared__ int      fIdx[16];

  const int  t       = threadIdx.x;
  const long rowbase = (long)blockIdx.x * 16;
  const int  wave    = t >> 5;
  const int  lane    = t & 31;

  // ---- Phase 1: global f16 residual tile -> LDS ----
#if __has_builtin(__builtin_amdgcn_tensor_load_to_lds) && defined(__gfx1250__)
  if (wave == 0) {
    // 2D Tensor DMA descriptor: 16 rows x 256 dwords, row stride 256 dwords,
    // LDS padding: +4 dwords after every 256 dwords -> matches lA's 520-half
    // row stride. data_size = 4 bytes.
    const unsigned long long ga =
        (unsigned long long)(const void*)(residh + rowbase * D_);
    const unsigned lds_off = (unsigned)(unsigned long long)(void*)&lA[0][0];
    u32x4 g0;
    g0[0] = 1u;                                    // count=1, user descriptor
    g0[1] = lds_off;                               // lds_addr
    g0[2] = (unsigned)(ga & 0xFFFFFFFFull);        // global_addr[31:0]
    g0[3] = (unsigned)((ga >> 32) & 0x1FFFFFFull)  // global_addr[56:32]
            | (2u << 30);                          // type = 2 ("image")
    i32x8 g1;
    g1[0] = (2 << 16)      // data_size = 4B
          | (1 << 20)      // pad_enable
          | (7 << 22)      // pad_interval: 256 dwords
          | (3 << 25);     // pad_amount: 4 dwords
    g1[1] = (256 << 16);   // tensor_dim0 = 256 (low 16)
    g1[2] = (16 << 16);    // tensor_dim0 hi = 0 | tensor_dim1 = 16 (low 16)
    g1[3] = (256 << 16);   // tensor_dim1 hi = 0 | tile_dim0 = 256
    g1[4] = 16;            // tile_dim1 = 16, tile_dim2 = 0
    g1[5] = 256;           // tensor_dim0_stride[31:0] = 256
    g1[6] = 0;
    g1[7] = 0;
    i32x4 gz4 = {0, 0, 0, 0};
    i32x8 gz8 = {0, 0, 0, 0, 0, 0, 0, 0};
    __builtin_amdgcn_tensor_load_to_lds(g0, g1, gz4, gz4, gz8, 0);
    __builtin_amdgcn_s_wait_tensorcnt(0);
  }
#else
  {
    const int r  = t >> 4;           // 0..15
    const int c0 = (t & 15) * 32;    // 0..480
    const _Float16* src = residh + (rowbase + r) * D_ + c0;
    v8h* dst = (v8h*)&lA[r][c0];
#pragma unroll
    for (int j = 0; j < 4; ++j) dst[j] = *(const v8h*)(src + j * 8);
  }
#endif
  __syncthreads();

  // ---- Phase 2: WMMA distance sweep + argmin ----
  const int col     = lane & 15;     // N (code) within tile / A row M
  const int halfSel = lane >> 4;     // lane half selects K sub-range

  float bestS[8];
  int   bestI[8];
#pragma unroll
  for (int r = 0; r < 8; ++r) { bestS[r] = 3.4e38f; bestI[r] = 0; }

  for (int kt = wave; kt < KTILES; kt += 8) {
    const int code = kt * 16 + col;
    const _Float16* brow = cbh + (long)code * D_;
    if (kt + 8 < KTILES)
      __builtin_prefetch(brow + 128 * D_, 0, 1);   // next tile's panel (L2)
    v8f c0 = {};
    v8f c1 = {};
#pragma unroll
    for (int kc = 0; kc < 16; kc += 2) {
      // A fragment (16x32 f16): lane half 0 -> K k0+0..7 & k0+16..23,
      //                         lane half 1 -> K k0+8..15 & k0+24..31, row=col
      // B fragment (32x16 f16): same K split as A (half 0 -> k0+0..7 & +16..23,
      //                         half 1 -> k0+8..15 & +24..31), column N = col.
      {
        const int k0 = kc * 32;
        v8h alo = *(const v8h*)&lA[col][k0 + halfSel * 8];
        v8h ahi = *(const v8h*)&lA[col][k0 + 16 + halfSel * 8];
        v16h a = __builtin_shufflevector(alo, ahi, 0, 1, 2, 3, 4, 5, 6, 7,
                                         8, 9, 10, 11, 12, 13, 14, 15);
        v8h blo = *(const v8h*)(brow + k0 + halfSel * 8);
        v8h bhi = *(const v8h*)(brow + k0 + 16 + halfSel * 8);
        v16h b = __builtin_shufflevector(blo, bhi, 0, 1, 2, 3, 4, 5, 6, 7,
                                         8, 9, 10, 11, 12, 13, 14, 15);
        c0 = __builtin_amdgcn_wmma_f32_16x16x32_f16(false, a, false, b,
                                                    (short)0, c0, false, false);
      }
      {
        const int k0 = (kc + 1) * 32;
        v8h alo = *(const v8h*)&lA[col][k0 + halfSel * 8];
        v8h ahi = *(const v8h*)&lA[col][k0 + 16 + halfSel * 8];
        v16h a = __builtin_shufflevector(alo, ahi, 0, 1, 2, 3, 4, 5, 6, 7,
                                         8, 9, 10, 11, 12, 13, 14, 15);
        v8h blo = *(const v8h*)(brow + k0 + halfSel * 8);
        v8h bhi = *(const v8h*)(brow + k0 + 16 + halfSel * 8);
        v16h b = __builtin_shufflevector(blo, bhi, 0, 1, 2, 3, 4, 5, 6, 7,
                                         8, 9, 10, 11, 12, 13, 14, 15);
        c1 = __builtin_amdgcn_wmma_f32_16x16x32_f16(false, a, false, b,
                                                    (short)0, c1, false, false);
      }
    }
    const v8f c = c0 + c1;
    const float en = enorm[code];
#pragma unroll
    for (int r = 0; r < 8; ++r) {            // rows M = r (+8 for upper half)
      const float s = fmaf(-2.0f, c[r], en);
      if (s < bestS[r]) { bestS[r] = s; bestI[r] = code; }  // ascending scan
    }
  }

  // butterfly min across the 16 lanes of each half (xor masks stay in-half)
#pragma unroll
  for (int r = 0; r < 8; ++r) {
    float s = bestS[r]; int i = bestI[r];
#pragma unroll
    for (int m = 1; m < 16; m <<= 1) {
      const float os = __shfl_xor(s, m, 32);
      const int   oi = __shfl_xor(i, m, 32);
      if (os < s || (os == s && oi < i)) { s = os; i = oi; }
    }
    bestS[r] = s; bestI[r] = i;
  }
  if (lane == 0 || lane == 16) {
    const int rbase = halfSel * 8;           // half 0 -> rows 0..7, half 1 -> 8..15
#pragma unroll
    for (int r = 0; r < 8; ++r) {
      wredS[wave][rbase + r] = bestS[r];
      wredI[wave][rbase + r] = bestI[r];
    }
  }
  __syncthreads();
  if (t < 16) {
    float s = wredS[0][t]; int i = wredI[0][t];
#pragma unroll
    for (int w = 1; w < 8; ++w) {
      const float os = wredS[w][t]; const int oi = wredI[w][t];
      if (os < s || (os == s && oi < i)) { s = os; i = oi; }
    }
    fIdx[t] = i;
  }
  __syncthreads();

  // ---- Phase 3: gather + residual / output / loss update ----
  {
    const int  r    = t >> 4;
    const int  c0   = (t & 15) * 32;
    const long g    = rowbase + r;
    const int  code = fIdx[r];
    const float* q  = cbf + (long)code * D_ + c0;
    float*     rp   = resid  + g * D_ + c0;
    _Float16*  rh   = residh + g * D_ + c0;
    float*     op   = outq   + g * D_ + c0;
    float acc = 0.0f;
#pragma unroll
    for (int j = 0; j < 8; ++j) {
      v4f qv = *(const v4f*)(q  + j * 4);
      v4f rv = *(const v4f*)(rp + j * 4);
      v4f ov = *(const v4f*)(op + j * 4);
      v4f dv = rv - qv;                      // new residual = r - q
      acc += dv.x * dv.x + dv.y * dv.y + dv.z * dv.z + dv.w * dv.w;
      *(v4f*)(rp + j * 4) = dv;
      *(v4f*)(op + j * 4) = ov + qv;         // quantized_out += q
      v4h hv = { (_Float16)dv.x, (_Float16)dv.y, (_Float16)dv.z, (_Float16)dv.w };
      *(v4h*)(rh + j * 4) = hv;
    }
#pragma unroll
    for (int m = 1; m < 32; m <<= 1) acc += __shfl_xor(acc, m, 32);
    if (lane == 0) atomicAdd(ssq, acc);
    if ((t & 15) == 0) idxout[g * L_ + layer] = (float)code;
  }
}

__global__ void finalize_kernel(float* __restrict__ outp,
                                const float* __restrict__ ssq) {
  if (threadIdx.x == 0 && blockIdx.x == 0)
    outp[NQ_] = COMMIT * (*ssq) / (float)NQ_;
}

// ---------------------------------------------------------------------------
// Host launcher
// ---------------------------------------------------------------------------
extern "C" void kernel_launch(void* const* d_in, const int* in_sizes, int n_in,
                              void* d_out, int out_size, void* d_ws, size_t ws_size,
                              hipStream_t stream) {
  (void)in_sizes; (void)n_in; (void)ws_size;
  const float* x  = (const float*)d_in[0];   // [B, W, D] f32
  const float* cb = (const float*)d_in[1];   // [L, K, D] f32
  float* out = (float*)d_out;                // [N*D] quantized + [1] loss + [N*L] idx

  char* ws = (char*)d_ws;
  size_t off = 0;
  float*    resid  = (float*)(ws + off);    off += (size_t)NQ_ * 4;
  _Float16* residh = (_Float16*)(ws + off); off += (size_t)NQ_ * 2;
  _Float16* cbh    = (_Float16*)(ws + off); off += (size_t)L_ * K_ * D_ * 2;
  float*    enorm  = (float*)(ws + off);    off += (size_t)L_ * K_ * 4;
  float*    ssq    = (float*)(ws + off);    off += 256;

  // 1) codebook f16 copies + ||e||^2
  prep_codebook<<<(L_ * K_) / 8, 256, 0, stream>>>(cb, cbh, enorm);

  // 2) residual <- x, zero outputs + loss accumulator
  const long initN = (out_size > NQ_) ? (long)out_size : NQ_;
  init_kernel<<<(int)((initN + 255) / 256), 256, 0, stream>>>(
      x, resid, residh, out, (long)out_size, ssq);

  // 3) eight cascaded VQ layers (sequential: residual dependence)
  float* idxout = out + NQ_ + 1;
  for (int l = 0; l < L_; ++l) {
    rvq_layer<<<ROWTILES, 256, 0, stream>>>(
        resid, residh,
        cbh + (size_t)l * K_ * D_,
        cb  + (size_t)l * K_ * D_,
        enorm + (size_t)l * K_,
        out, idxout, ssq, l);
  }

  // 4) loss = 0.25 * sum((r-q)^2) / (N*D)
  finalize_kernel<<<1, 1, 0, stream>>>(out, ssq);
}
